// model2_53953379172891
// MI455X (gfx1250) — matrix operand order repaired and checked by reading, not compiled, not run on hardware
//
#include <hip/hip_runtime.h>
#include <hip/hip_bf16.h>
#include <stdint.h>

typedef __attribute__((ext_vector_type(16))) _Float16 v16h;
typedef __attribute__((ext_vector_type(8)))  float    v8f;

#define HMM_B 256
#define HMM_T 1024
#define HMM_D 88
#define HMM_K 16
#define NFEAT 264          // y(88) | yprev(88) | y*yprev(88)
#define NFEAT_PAD 288      // 9 * 32
#define NKSTEP 9

// ---------------- workspace layout (bytes) ----------------
#define WS_BFRAG_OFF 16777216
#define WS_CK_OFF    16786432
#define WS_PRIOR_OFF 16786496
#define WS_LL_OFF    16786560

#if defined(__gfx1250__) && __has_builtin(__builtin_amdgcn_global_load_async_to_lds_b128)
#define HAVE_ASYNC_LDS 1
typedef int v4i_ __attribute__((vector_size(16)));
typedef __attribute__((address_space(1))) v4i_* gptr128_t;
typedef __attribute__((address_space(3))) v4i_* lptr128_t;
#endif

__device__ __forceinline__ void wait_async_lds() {
#if defined(HAVE_ASYNC_LDS)
#if __has_builtin(__builtin_amdgcn_s_wait_asynccnt)
    __builtin_amdgcn_s_wait_asynccnt(0);
#else
    asm volatile("s_wait_asynccnt 0x0" ::: "memory");
#endif
#endif
}

// ============================================================
// Kernel 1: build weight fragments, C[k], and prior scalar
// ============================================================
__global__ __launch_bounds__(256) void k_prep(const float* __restrict__ px,
                                              const float* __restrict__ py,
                                              unsigned* __restrict__ bfrag,
                                              float* __restrict__ ck,
                                              float* __restrict__ prior)
{
    __shared__ float Wsh[NFEAT_PAD * HMM_K];   // 18,432 B
    __shared__ float red[256];
    const int tid = threadIdx.x;

    // ---- weight matrix W[f][k] (f32 in LDS) ----
    for (int idx = tid; idx < NFEAT_PAD * HMM_K; idx += 256) {
        int f = idx >> 4, k = idx & 15;
        float w = 0.f;
        if (f < NFEAT) {
            if (f < HMM_D) {                       // y   : logit(p_{k,0,d})
                float p = py[(k * 2 + 0) * HMM_D + f];
                w = __logf(p) - log1pf(-p);
            } else if (f < 2 * HMM_D) {            // yp  : log(1-p1) - log(1-p0)
                int d = f - HMM_D;
                float p1 = py[(k * 2 + 1) * HMM_D + d];
                float p0 = py[(k * 2 + 0) * HMM_D + d];
                w = log1pf(-p1) - log1pf(-p0);
            } else {                               // y*yp: logit(p1) - logit(p0)
                int d = f - 2 * HMM_D;
                float p1 = py[(k * 2 + 1) * HMM_D + d];
                float p0 = py[(k * 2 + 0) * HMM_D + d];
                w = (__logf(p1) - log1pf(-p1)) - (__logf(p0) - log1pf(-p0));
            }
        }
        Wsh[idx] = w;
    }

    // ---- beta prior partials + dirichlet partial ----
    float local = 0.f;
    for (int i = tid; i < HMM_K * 2 * HMM_D; i += 256) {
        float p = py[i];
        local += -0.9f * __logf(p) - 0.1f * log1pf(-p);
    }
    {
        int i = tid;
        int r = i >> 4, c = i & 15;
        if (r != c) local += -0.9f * __logf(px[i]);
    }

    // ---- C[k] = sum_d log(1 - p_{k,0,d}) ----
    if (tid < HMM_K) {
        float c = 0.f;
        for (int d = 0; d < HMM_D; ++d) c += log1pf(-py[(tid * 2 + 0) * HMM_D + d]);
        ck[tid] = c;
    }

    red[tid] = local;
    __syncthreads();
    for (int s = 128; s > 0; s >>= 1) {
        if (tid < s) red[tid] += red[tid + s];
        __syncthreads();
    }
    if (tid == 0) {
        const double g25 = 0.28468287047291915;   // gammaln(2.5)
        const double g01 = 2.2527126517342055;    // gammaln(0.1)
        const double g09 = 0.06637623973474297;   // gammaln(0.9)
        double dir_c  = 16.0 * (g25 - 15.0 * g01);
        double beta_c = -(double)(HMM_K * 2 * HMM_D) * (g01 + g09);
        prior[0] = (float)((double)red[0] + dir_c + beta_c);
    }

    // ---- pack B fragments: lane l -> col l%16 ; VGPR v -> K-rows (table order) k, k+1 ----
    for (int idx = tid; idx < NKSTEP * 32 * 8; idx += 256) {
        int kk  = idx >> 8;
        int rem = idx & 255;
        int l   = rem >> 3;
        int v   = rem & 7;
        int krow = kk * 32 + (l >> 4) * 8 + v * 2 + ((v >> 2) << 3);
        int n    = l & 15;
        union { _Float16 h[2]; unsigned u; } pk;
        pk.h[0] = (_Float16)Wsh[krow * HMM_K + n];
        pk.h[1] = (_Float16)Wsh[(krow + 1) * HMM_K + n];
        bfrag[idx] = pk.u;
    }
}

// ============================================================
// Kernel 2: emission GEMM via v_wmma_f32_16x16x32_f16
//   block = 128 threads (4 waves); covers one b, 64 t-rows.
//   Stage A: async-copy raw f32 rows to LDS.
//   Stage B: build packed f16 feature rows feat[64][288].
//   Stage C: A = 2x ds_load_b128 per k-step; 9 WMMAs per tile,
//            dual accumulators to break WMMA->WMMA RAW hazards.
// ============================================================
__global__ __launch_bounds__(128) void k_emit(const float* __restrict__ seq,
                                              const unsigned* __restrict__ bfrag,
                                              const float* __restrict__ ck,
                                              float* __restrict__ emit)
{
    __shared__ __align__(16) float ys[65 * HMM_D];                // 22,880 B
    __shared__ __align__(16) unsigned short feat[64 * NFEAT_PAD]; // 36,864 B
    const int tid = threadIdx.x;
    const int b  = blockIdx.x >> 4;
    const int t0 = (blockIdx.x & 15) << 6;

    // ---- Stage A: raw rows t0-1 .. t0+63 into ys (row -1 = zeros) ----
    {
        float*       dst;
        const float* gsrc;
        int          nfloat;
        if (t0 == 0) {
            for (int d = tid; d < HMM_D; d += 128) ys[d] = 0.f;
            dst    = ys + HMM_D;
            gsrc   = seq + (size_t)b * HMM_T * HMM_D;
            nfloat = 64 * HMM_D;                 // 22,528 B, 16B-aligned
        } else {
            dst    = ys;
            gsrc   = seq + ((size_t)b * HMM_T + t0 - 1) * HMM_D;
            nfloat = 65 * HMM_D;                 // 22,880 B, 16B-aligned
        }
#if defined(HAVE_ASYNC_LDS)
        for (int i = tid * 4; i < nfloat; i += 128 * 4) {
            __builtin_amdgcn_global_load_async_to_lds_b128(
                (gptr128_t)(gsrc + i), (lptr128_t)(dst + i), 0, 0);
        }
        wait_async_lds();
#else
        for (int i = tid * 4; i < nfloat; i += 128 * 4) {
            float4 v = *(const float4*)(gsrc + i);
            *(float4*)(dst + i) = v;
        }
#endif
    }
    __syncthreads();

    // ---- Stage B: packed f16 feature rows ----
    for (int i = tid; i < 64 * (HMM_D / 2); i += 128) {
        int r  = i / (HMM_D / 2);
        int d0 = (i - r * (HMM_D / 2)) * 2;
        float y0  = ys[(r + 1) * HMM_D + d0],     y1  = ys[(r + 1) * HMM_D + d0 + 1];
        float yp0 = ys[r * HMM_D + d0],           yp1 = ys[r * HMM_D + d0 + 1];
        union { _Float16 h[2]; unsigned u; } pk;
        unsigned* fr = (unsigned*)(feat + r * NFEAT_PAD);
        pk.h[0] = (_Float16)y0;         pk.h[1] = (_Float16)y1;
        fr[d0 >> 1] = pk.u;
        pk.h[0] = (_Float16)yp0;        pk.h[1] = (_Float16)yp1;
        fr[(HMM_D + d0) >> 1] = pk.u;
        pk.h[0] = (_Float16)(y0 * yp0); pk.h[1] = (_Float16)(y1 * yp1);
        fr[(2 * HMM_D + d0) >> 1] = pk.u;
    }
    // zero-pad features 264..287
    for (int i = tid; i < 64 * ((NFEAT_PAD - NFEAT) / 2); i += 128) {
        int r = i / 12, c = i - r * 12;
        ((unsigned*)(feat + r * NFEAT_PAD))[(NFEAT >> 1) + c] = 0u;
    }
    __syncthreads();

    // ---- Stage C: WMMA, two independent accumulator chains ----
    const int wave = tid >> 5;
    const int lane = tid & 31;
    const int half = lane >> 4;
    const int mrow = lane & 15;
    const int row  = wave * 16 + mrow;            // t = t0 + row
    const uint4* fp = (const uint4*)(feat + row * NFEAT_PAD);

    v8f acc0 = {};
    v8f acc1 = {};
#pragma unroll
    for (int kk = 0; kk < NKSTEP; ++kk) {
        union { v16h v; uint4 q[2]; } A;
        A.q[0] = fp[kk * 4 + half];        // features kk*32 + half*8 .. +7
        A.q[1] = fp[kk * 4 + 2 + half];    // features kk*32+16+half*8 .. +7

        union { v16h v; uint4 q[2]; } Bf;
        const uint4* bq = (const uint4*)(bfrag + (size_t)(kk * 32 + lane) * 8);
        Bf.q[0] = bq[0];
        Bf.q[1] = bq[1];

        if (kk & 1)
            acc1 = __builtin_amdgcn_wmma_f32_16x16x32_f16(
                false, A.v, false, Bf.v, (short)0, acc1, false, false);
        else
            acc0 = __builtin_amdgcn_wmma_f32_16x16x32_f16(
                false, A.v, false, Bf.v, (short)0, acc0, false, false);
    }
    v8f acc = acc0 + acc1;

    const float cc = ck[lane & 15];
    const int tb = t0 + wave * 16;
#pragma unroll
    for (int r = 0; r < 8; ++r) {
        int t = tb + r + half * 8;  // D layout: VGPR r, lanes 16-31 -> M = r+8
        emit[((size_t)(b * HMM_T + t)) * HMM_K + (lane & 15)] = acc[r] + cc;
    }
}

// ============================================================
// Kernel 3: scaled forward recursion (linear domain, per-step renorm)
//   1 wave = 2 batches; 16 lanes = 16 states.
//   alpha gathered via LDS (1 store + 4 b128 loads per step).
// ============================================================
__global__ __launch_bounds__(32) void k_forward(const float* __restrict__ px,
                                                const int* __restrict__ lengths,
                                                const float* __restrict__ emit,
                                                float* __restrict__ ll)
{
    __shared__ __align__(16) float as_[32];
    const int lane = threadIdx.x;
    const int grp  = lane >> 4;
    const int j    = lane & 15;
    const int b    = blockIdx.x * 2 + grp;
    const int base = grp << 4;

    float pxc[16];
#pragma unroll
    for (int i = 0; i < 16; ++i) pxc[i] = px[i * HMM_K + j];  // column j

    float a    = (j == 0) ? 1.f : 0.f;
    float logc = 0.f;
    const int len = lengths[b];
    const float* eb = emit + (size_t)b * HMM_T * HMM_K;

    for (int t = 0; t < HMM_T; ++t) {
        float e = eb[t * HMM_K + j];
        // publish alpha, gather 16 values (wave-internal LDS is in-order)
        as_[lane] = a;
        __builtin_amdgcn_wave_barrier();
        const float4* ap = (const float4*)(as_ + base);
        float4 v0 = ap[0], v1 = ap[1], v2 = ap[2], v3 = ap[3];
        float tmp = 0.f;
        tmp = fmaf(v0.x, pxc[0],  tmp); tmp = fmaf(v0.y, pxc[1],  tmp);
        tmp = fmaf(v0.z, pxc[2],  tmp); tmp = fmaf(v0.w, pxc[3],  tmp);
        tmp = fmaf(v1.x, pxc[4],  tmp); tmp = fmaf(v1.y, pxc[5],  tmp);
        tmp = fmaf(v1.z, pxc[6],  tmp); tmp = fmaf(v1.w, pxc[7],  tmp);
        tmp = fmaf(v2.x, pxc[8],  tmp); tmp = fmaf(v2.y, pxc[9],  tmp);
        tmp = fmaf(v2.z, pxc[10], tmp); tmp = fmaf(v2.w, pxc[11], tmp);
        tmp = fmaf(v3.x, pxc[12], tmp); tmp = fmaf(v3.y, pxc[13], tmp);
        tmp = fmaf(v3.z, pxc[14], tmp); tmp = fmaf(v3.w, pxc[15], tmp);
        // m = max_j e within 16-lane group
        float m = e;
        m = fmaxf(m, __shfl_xor(m, 1, 32));
        m = fmaxf(m, __shfl_xor(m, 2, 32));
        m = fmaxf(m, __shfl_xor(m, 4, 32));
        m = fmaxf(m, __shfl_xor(m, 8, 32));
        float an = tmp * __expf(e - m);
        float s = an;
        s = fmaxf(s, __shfl_xor(s, 1, 32));
        s = fmaxf(s, __shfl_xor(s, 2, 32));
        s = fmaxf(s, __shfl_xor(s, 4, 32));
        s = fmaxf(s, __shfl_xor(s, 8, 32));
        bool upd = (t < len);
        float a2 = an * __builtin_amdgcn_rcpf(s);  // renorm: approx rcp is fine
        float l2 = logc + m + __logf(s);
        a    = upd ? a2 : a;
        logc = upd ? l2 : logc;
    }
    float sa = a;
    sa += __shfl_xor(sa, 1, 32);
    sa += __shfl_xor(sa, 2, 32);
    sa += __shfl_xor(sa, 4, 32);
    sa += __shfl_xor(sa, 8, 32);
    if (j == 0) ll[b] = logc + __logf(sa);
}

// ============================================================
// Kernel 4: deterministic final reduction + priors
// ============================================================
__global__ __launch_bounds__(256) void k_final(const float* __restrict__ ll,
                                               const float* __restrict__ prior,
                                               float* __restrict__ out)
{
    __shared__ float red[256];
    const int tid = threadIdx.x;
    red[tid] = ll[tid];
    __syncthreads();
    for (int s = 128; s > 0; s >>= 1) {
        if (tid < s) red[tid] += red[tid + s];
        __syncthreads();
    }
    if (tid == 0) out[0] = red[0] + prior[0];
}

// ============================================================
extern "C" void kernel_launch(void* const* d_in, const int* in_sizes, int n_in,
                              void* d_out, int out_size, void* d_ws, size_t ws_size,
                              hipStream_t stream)
{
    const float* seq = (const float*)d_in[0];   // [B,T,D] f32 (0/1)
    const float* px  = (const float*)d_in[1];   // [K,K]   f32
    const float* py  = (const float*)d_in[2];   // [K,2,D] f32
    const int*   len = (const int*)d_in[3];     // [B]     i32

    char* ws = (char*)d_ws;
    float*    emit  = (float*)ws;
    unsigned* bfrag = (unsigned*)(ws + WS_BFRAG_OFF);
    float*    ck    = (float*)(ws + WS_CK_OFF);
    float*    prior = (float*)(ws + WS_PRIOR_OFF);
    float*    ll    = (float*)(ws + WS_LL_OFF);
    float*    out   = (float*)d_out;

    k_prep<<<1, 256, 0, stream>>>(px, py, bfrag, ck, prior);
    k_emit<<<HMM_B * (HMM_T / 64), 128, 0, stream>>>(seq, bfrag, ck, emit);
    k_forward<<<HMM_B / 2, 32, 0, stream>>>(px, len, emit, ll);
    k_final<<<1, 256, 0, stream>>>(ll, prior, out);
}
